// AC_40458591929191
// MI455X (gfx1250) — compile-verified, matrix operand order checked by reading, not run on hardware
//
#include <hip/hip_runtime.h>
#include <math.h>

// ---------------- problem constants ----------------
constexpr int BB   = 2;
constexpr int HGT  = 128;
constexpr int WID  = 128;
constexpr int NTOK = HGT * WID;        // 16384
constexpr int DIM  = 192;
constexpr int HEADS= 6;
constexpr int HD   = DIM / HEADS;      // 32
constexpr int WS2  = 64;
constexpr int NW   = NTOK / WS2;       // 256 windows per batch
constexpr int NWIN = BB * NW;          // 512 windows total
constexpr int BH   = BB * HEADS;       // 12
constexpr int QKVC = 3 * DIM;          // 576
constexpr int MROWS= BB * NTOK;        // 32768

typedef _Float16 v16h __attribute__((ext_vector_type(16)));
typedef float    v8f  __attribute__((ext_vector_type(8)));

union Frag16 { v16h h; uint4 q[2]; };

// Load a 16-half WMMA fragment (A-layout) from a row-major matrix where the
// 16 halves of lane L are K-contiguous: elements 0..7 at k0+8*hi, 8..15 at
// k0+16+8*hi. Requires 16-byte alignment (all our lds/ld strides guarantee it).
__device__ __forceinline__ v16h load_frag(const _Float16* __restrict__ base,
                                          int row, int ld, int k0, int hi)
{
    Frag16 f;
    const _Float16* p = base + (size_t)row * ld + k0 + (hi << 3);
    f.q[0] = *(const uint4*)(p);
    f.q[1] = *(const uint4*)(p + 16);
    return f.h;
}

__device__ __forceinline__ v16h load_frag_lds(const _Float16* base,
                                              int row, int ld, int k0, int hi)
{
    Frag16 f;
    const _Float16* p = base + row * ld + k0 + (hi << 3);
    f.q[0] = *(const uint4*)(p);
    f.q[1] = *(const uint4*)(p + 16);
    return f.h;
}

// ============================================================
// Generic f16 WMMA GEMM: C(MxN,f32) = A(MxK) * BT(NxK)^T [+ bias]
// A row-major lda, BT row-major ldb (i.e. B stored N-major), C row-major ldc.
// M,N multiples of 16; K multiple of 32. One 16x16 tile per wave.
// Both operands load as 2x b128 per fragment (K-contiguous).
// ============================================================
__global__ void gemm_f16_wmma_kernel(const _Float16* __restrict__ A,
                                     const _Float16* __restrict__ BT,
                                     const float* __restrict__ bias,
                                     float* __restrict__ C,
                                     int M, int N, int K,
                                     int lda, int ldb, int ldc)
{
    const int lane = threadIdx.x & 31;
    const int wave = threadIdx.x >> 5;
    const int ntn  = N >> 4;
    const int total = (M >> 4) * ntn;
    const int tile = blockIdx.x * 4 + wave;
    if (tile >= total) return;                   // whole-wave uniform exit
    const int tm = tile / ntn, tn = tile % ntn;
    const int m0 = tm << 4, n0 = tn << 4;
    const int am = lane & 15;                    // A row / BT row / C col
    const int hi = lane >> 4;

    v8f acc = {};
    for (int k0 = 0; k0 < K; k0 += 32) {
        v16h a = load_frag(A,  m0 + am, lda, k0, hi);
        v16h b = load_frag(BT, n0 + am, ldb, k0, hi);
        acc = __builtin_amdgcn_wmma_f32_16x16x32_f16(false, a, false, b,
                                                     (short)0, acc, false, false);
    }
    const float bv = bias ? bias[n0 + am] : 0.0f;
#pragma unroll
    for (int r = 0; r < 8; ++r) {
        const int row = m0 + r + (hi << 3);
        C[row * ldc + (n0 + am)] = acc[r] + bv;
    }
}

// ============================================================
// Converters
// ============================================================
__global__ void cvt_f32_f16_kernel(const float* __restrict__ src,
                                   _Float16* __restrict__ dst, int n)
{
    int i = blockIdx.x * 256 + threadIdx.x;
    if (i < n) dst[i] = (_Float16)src[i];
}

// ============================================================
// Clustering prep: q_pre_h[bh][d][n], qn_h[bh][n][d]  (f16)
// One token per 32 lanes; lane = d.
// ============================================================
__global__ void extract_q_kernel(const float* __restrict__ qkv,
                                 _Float16* __restrict__ qpre,
                                 _Float16* __restrict__ qn)
{
    const int gtok = blockIdx.x * 4 + (threadIdx.x >> 5);
    const int lane = threadIdx.x & 31;                     // d
    if (gtok >= BH * NTOK) return;
    const int bh = gtok / NTOK, n = gtok % NTOK;
    const int b = bh / HEADS, hh = bh % HEADS;
    const float v = qkv[((size_t)b * NTOK + n) * QKVC + hh * HD + lane];
    float ss = v * v;
#pragma unroll
    for (int off = 16; off >= 1; off >>= 1) ss += __shfl_xor(ss, off, 32);
    const float nrm = fmaxf(sqrtf(ss), 1e-12f);
    qpre[((size_t)bh * HD + lane) * NTOK + n] = (_Float16)v;
    qn[((size_t)bh * NTOK + n) * HD + lane]   = (_Float16)(v / nrm);
}

// Initial centroids, stored N-major for the scores GEMM:
// qavg[bh][c][d] = l2n_d(mean over window tokens).  grid = BH*NW, block = 32.
__global__ void cluster_init_kernel(const float* __restrict__ qkv,
                                    _Float16* __restrict__ qavg)
{
    const int bc = blockIdx.x;
    const int bh = bc / NW, cidx = bc % NW;
    const int d = threadIdx.x;
    const int b = bh / HEADS, hh = bh % HEADS;
    float sum = 0.0f;
    for (int s = 0; s < WS2; ++s)
        sum += qkv[((size_t)b * NTOK + (cidx * WS2 + s)) * QKVC + hh * HD + d];
    const float mean = sum * (1.0f / WS2);
    float ss = mean * mean;
#pragma unroll
    for (int off = 16; off >= 1; off >>= 1) ss += __shfl_xor(ss, off, 32);
    const float nrm = fmaxf(sqrtf(ss), 1e-12f);
    qavg[((size_t)bh * NW + cidx) * HD + d] = (_Float16)(mean / nrm);
}

// softmax(scores*100) rows of length 256 -> TRANSPOSED f16 probs [c][n]
// (N-major B operand for the centroid-update GEMM). One wave per row.
__global__ void softmax_rows_kernel(const float* __restrict__ scores,
                                    _Float16* __restrict__ probs_t)
{
    const int row = blockIdx.x * 8 + (threadIdx.x >> 5);
    const int lane = threadIdx.x & 31;
    if (row >= NTOK) return;
    const float* srow = scores + (size_t)row * NW;
    float v[8];
    float mx = -3.0e38f;
#pragma unroll
    for (int j = 0; j < 8; ++j) { v[j] = srow[lane + j * 32] * 100.0f; mx = fmaxf(mx, v[j]); }
#pragma unroll
    for (int off = 16; off >= 1; off >>= 1) mx = fmaxf(mx, __shfl_xor(mx, off, 32));
    float sum = 0.0f;
#pragma unroll
    for (int j = 0; j < 8; ++j) { v[j] = expf(v[j] - mx); sum += v[j]; }
#pragma unroll
    for (int off = 16; off >= 1; off >>= 1) sum += __shfl_xor(sum, off, 32);
    const float inv = 1.0f / sum;
#pragma unroll
    for (int j = 0; j < 8; ++j)
        probs_t[(size_t)(lane + j * 32) * NTOK + row] = (_Float16)(v[j] * inv);
}

// l2-normalize columns of qacc (HD x NW, f32) -> qavg slice, N-major [c][d]
__global__ void norm_qavg_kernel(const float* __restrict__ qacc,
                                 _Float16* __restrict__ qavg_bh)
{
    const int c = blockIdx.x * 64 + threadIdx.x;
    if (c >= NW) return;
    float ss = 0.0f;
    for (int d = 0; d < HD; ++d) { float t = qacc[d * NW + c]; ss += t * t; }
    const float inv = 1.0f / fmaxf(sqrtf(ss), 1e-12f);
    for (int d = 0; d < HD; ++d)
        qavg_bh[c * HD + d] = (_Float16)(qacc[d * NW + c] * inv);
}

__global__ void ratio_kernel(const float* __restrict__ scores,
                             float* __restrict__ ratio_bh)
{
    const int n = blockIdx.x * 256 + threadIdx.x;
    if (n < NTOK)
        ratio_bh[n] = (scores[(size_t)n * NW + 0] + 1.0f) /
                      (scores[(size_t)n * NW + 1] + 1.0f);
}

// ============================================================
// Stable-ascending bitonic argsort of 16384 keys per bh (in LDS).
// Key = sortable-float(ratio) << 32 | index  => stable order.
// grid = BH, block = 1024, dynamic LDS = 16384*8 bytes.
// ============================================================
__global__ void sort_kernel(const float* __restrict__ ratio,
                            int* __restrict__ qidx)
{
    extern __shared__ unsigned long long keys[];
    const int bh = blockIdx.x, tid = threadIdx.x;
    const float* r = ratio + (size_t)bh * NTOK;
    for (int i = tid; i < NTOK; i += 1024) {
        unsigned u = __float_as_uint(r[i]);
        u = (u & 0x80000000u) ? ~u : (u | 0x80000000u);
        keys[i] = ((unsigned long long)u << 32) | (unsigned)i;
    }
    __syncthreads();
    for (int k = 2; k <= NTOK; k <<= 1) {
        for (int j = k >> 1; j > 0; j >>= 1) {
            for (int i = tid; i < NTOK; i += 1024) {
                const int ixj = i ^ j;
                if (ixj > i) {
                    const bool up = ((i & k) == 0);
                    unsigned long long a = keys[i], b = keys[ixj];
                    if ((a > b) == up) { keys[i] = b; keys[ixj] = a; }
                }
            }
            __syncthreads();
        }
    }
    for (int i = tid; i < NTOK; i += 1024)
        qidx[(size_t)bh * NTOK + i] = (int)(keys[i] & 0xFFFFFFFFu);
}

__global__ void invperm_kernel(const int* __restrict__ qidx,
                               int* __restrict__ qrev)
{
    const int i = blockIdx.x * 256 + threadIdx.x;
    if (i < BH * NTOK) {
        const int bh = i >> 14, j = i & (NTOK - 1);
        qrev[(size_t)bh * NTOK + qidx[i]] = j;
    }
}

// ============================================================
// Gather permuted q/k/v windows (heads merged into channels), f16.
// win[p][w][s][c] = qkv[b][ q_idx[b,c/32, a*64+s] ][p*192 + c]
// ============================================================
__global__ void gather_windows_kernel(const float* __restrict__ qkv,
                                      const int* __restrict__ qidx,
                                      _Float16* __restrict__ win)
{
    const long long i = (long long)blockIdx.x * 256 + threadIdx.x;
    const long long tot = 3LL * NWIN * WS2 * DIM;
    if (i >= tot) return;
    const int c = (int)(i % DIM);
    long long rest = i / DIM;
    const int s = (int)(rest % WS2); rest /= WS2;
    const int w = (int)(rest % NWIN);
    const int p = (int)(rest / NWIN);
    const int b = w >> 8, a = w & (NW - 1);
    const int hh = c >> 5;
    const int tok = qidx[((size_t)(b * HEADS + hh)) * NTOK + a * WS2 + s];
    win[i] = (_Float16)qkv[((size_t)b * NTOK + tok) * QKVC + p * DIM + c];
}

// ============================================================
// Windowed attention: per window (64 tokens, 192 ch, heads merged, no scale)
//   S = Q K^T (WMMA, f32 regs) -> row softmax -> P(f16, LDS) -> O = P V (WMMA)
// block = 128 threads (4 waves, 16 query rows each), grid = 512 windows.
// K tiles kept row-major (already N-major for S); V staged transposed so the
// O=PV B-fragments are K-contiguous b128 loads too.
// ============================================================
__global__ void attn_win_kernel(const _Float16* __restrict__ Wq,
                                const _Float16* __restrict__ Wk,
                                const _Float16* __restrict__ Wv,
                                float* __restrict__ Out)
{
    __shared__ __align__(16) _Float16 Ks[WS2 * DIM];   // [s][c]  24.0 KB
    __shared__ __align__(16) _Float16 Vst[DIM * WS2];  // [c][s]  24.0 KB
    __shared__ __align__(16) _Float16 Pls[WS2 * 72];   // stride 72 -> 144B rows

    const int w = blockIdx.x;
    const int tid = threadIdx.x;
    const size_t base = (size_t)w * WS2 * DIM;

    // stage K (vectorized, layout-preserving) and V (transposed)
    {
        const uint4* k4 = (const uint4*)(Wk + base);
        uint4* ks4 = (uint4*)Ks;
        for (int i = tid; i < WS2 * DIM / 8; i += 128) ks4[i] = k4[i];
        for (int i = tid; i < WS2 * DIM; i += 128) {
            const int s = i / DIM, c = i % DIM;
            Vst[c * WS2 + s] = Wv[base + i];
        }
    }
    __syncthreads();

    const int lane = tid & 31, wv = tid >> 5;
    const int m0 = wv << 4;                  // 16-row strip per wave
    const int am = lane & 15, hi = lane >> 4;
    const _Float16* q = Wq + base;

    // ---- S = Q K^T : 4 n-tiles of 16x16, K=192 ----
    v8f s[4] = {};
    for (int k0 = 0; k0 < DIM; k0 += 32) {
        v16h a = load_frag(q, m0 + am, DIM, k0, hi);
#pragma unroll
        for (int tn = 0; tn < 4; ++tn) {
            v16h b = load_frag_lds(Ks, tn * 16 + am, DIM, k0, hi);
            s[tn] = __builtin_amdgcn_wmma_f32_16x16x32_f16(false, a, false, b,
                                                           (short)0, s[tn], false, false);
        }
    }

    // ---- row softmax (rows live at r + 8*hi across 16-lane halves) ----
#pragma unroll
    for (int r = 0; r < 8; ++r) {
        float mx = -3.0e38f;
#pragma unroll
        for (int tn = 0; tn < 4; ++tn) mx = fmaxf(mx, s[tn][r]);
#pragma unroll
        for (int off = 8; off >= 1; off >>= 1) mx = fmaxf(mx, __shfl_xor(mx, off, 32));
        float p[4], sum = 0.0f;
#pragma unroll
        for (int tn = 0; tn < 4; ++tn) { p[tn] = expf(s[tn][r] - mx); sum += p[tn]; }
#pragma unroll
        for (int off = 8; off >= 1; off >>= 1) sum += __shfl_xor(sum, off, 32);
        const float inv = 1.0f / sum;
        const int row = m0 + r + (hi << 3);
#pragma unroll
        for (int tn = 0; tn < 4; ++tn)
            Pls[row * 72 + tn * 16 + am] = (_Float16)(p[tn] * inv);
    }
    __syncthreads();

    // ---- O = P V : 12 n-tiles, K=64 ----
    for (int nt = 0; nt < 12; ++nt) {
        v8f o = {};
#pragma unroll
        for (int k0 = 0; k0 < WS2; k0 += 32) {
            v16h a = load_frag_lds(Pls, m0 + am, 72, k0, hi);
            v16h b = load_frag_lds(Vst, nt * 16 + am, WS2, k0, hi);
            o = __builtin_amdgcn_wmma_f32_16x16x32_f16(false, a, false, b,
                                                       (short)0, o, false, false);
        }
#pragma unroll
        for (int r = 0; r < 8; ++r)
            Out[((size_t)w * WS2 + m0 + r + (hi << 3)) * DIM + nt * 16 + am] = o[r];
    }
}

// ============================================================
// Fused: x_pre[b,n,c] = gather(attn_out via reshape chain + inverse perm)
//                     + depthwise3x3(v) + conv_b
// (v gather+inverse-gather cancels -> conv reads v-slice of qkv directly)
// ============================================================
__global__ void conv_merge_kernel(const float* __restrict__ qkv,
                                  const float* __restrict__ attn_out,
                                  const int* __restrict__ qrev,
                                  const float* __restrict__ convw,
                                  const float* __restrict__ convb,
                                  float* __restrict__ xpre)
{
    const int i = blockIdx.x * 256 + threadIdx.x;
    if (i >= BB * NTOK * DIM) return;
    const int c = i % DIM;
    const int n = (i / DIM) % NTOK;
    const int b = i / (DIM * NTOK);

    // attention-output gather (faithful reshape-chain composition)
    const int hh = c >> 5;
    const int t = qrev[((size_t)(b * HEADS + hh)) * NTOK + n];
    const int a = t >> 6, rr = t & 63;
    const int wdw = b * NW + a;
    const int idx = rr * DIM + c;
    const int cc = idx >> 6, s = idx & 63;
    const float aval = attn_out[((size_t)(wdw * WS2 + s)) * DIM + cc];

    // depthwise 3x3 SAME conv on v (channel c)
    const int y = n >> 7, x = n & (WID - 1);
    float acc = convb[c];
#pragma unroll
    for (int ky = 0; ky < 3; ++ky) {
        const int yy = y + ky - 1;
        if (yy < 0 || yy >= HGT) continue;
#pragma unroll
        for (int kx = 0; kx < 3; ++kx) {
            const int xx = x + kx - 1;
            if (xx < 0 || xx >= WID) continue;
            acc += convw[c * 9 + ky * 3 + kx] *
                   qkv[((size_t)b * NTOK + (yy * WID + xx)) * QKVC + 2 * DIM + c];
        }
    }
    xpre[i] = aval + acc;
}

// ============================================================
// Host side
// ============================================================
static inline int gemm_blocks(int M, int N) {
    const int tiles = (M >> 4) * (N >> 4);
    return (tiles + 3) >> 2;
}

extern "C" void kernel_launch(void* const* d_in, const int* in_sizes, int n_in,
                              void* d_out, int out_size, void* d_ws, size_t ws_size,
                              hipStream_t stream)
{
    const float* x      = (const float*)d_in[0];
    const float* qkv_w  = (const float*)d_in[1];
    const float* qkv_b  = (const float*)d_in[2];
    const float* proj_w = (const float*)d_in[3];
    const float* proj_b = (const float*)d_in[4];
    const float* conv_w = (const float*)d_in[5];
    const float* conv_b = (const float*)d_in[6];
    float* out = (float*)d_out;

    char* ws = (char*)d_ws;
    size_t off = 0;
    auto alloc = [&](size_t bytes) { char* p = ws + off; off += (bytes + 255) & ~(size_t)255; return p; };

    float*    qkv     = (float*)   alloc((size_t)MROWS * QKVC * 4);   // 75.5 MB
    _Float16* xh      = (_Float16*)alloc((size_t)MROWS * DIM * 2);    // 12.6 MB
    _Float16* wh      = (_Float16*)alloc((size_t)QKVC * DIM * 2);     // 0.22 MB
    _Float16* qpre    = (_Float16*)alloc((size_t)BH * HD * NTOK * 2); // 12.6 MB
    _Float16* qn      = (_Float16*)alloc((size_t)BH * NTOK * HD * 2); // 12.6 MB
    _Float16* qavg    = (_Float16*)alloc((size_t)BH * NW * HD * 2);   // N-major [bh][c][d]
    float*    qacc    = (float*)   alloc((size_t)HD * NW * 4);
    float*    scores  = (float*)   alloc((size_t)NTOK * NW * 4);      // 16.8 MB (reused per bh)
    _Float16* probs_t = (_Float16*)alloc((size_t)NW * NTOK * 2);      //  8.4 MB, [c][n]
    float*    ratio   = (float*)   alloc((size_t)BH * NTOK * 4);
    int*      qidx    = (int*)     alloc((size_t)BH * NTOK * 4);
    int*      qrev    = (int*)     alloc((size_t)BH * NTOK * 4);
    _Float16* win     = (_Float16*)alloc(3LL * NWIN * WS2 * DIM * 2); // 37.7 MB
    float*    attn    = (float*)   alloc((size_t)NWIN * WS2 * DIM * 4); // 25.2 MB
    float*    xpre    = (float*)   alloc((size_t)MROWS * DIM * 4);    // 25.2 MB

    // 1) QKV projection: qkv = x @ qkv_w^T + qkv_b.
    //    qkv_w is (N=576, K=192) row-major == the N-major B operand directly.
    cvt_f32_f16_kernel<<<(MROWS * DIM + 255) / 256, 256, 0, stream>>>(x, xh, MROWS * DIM);
    cvt_f32_f16_kernel<<<(QKVC * DIM + 255) / 256, 256, 0, stream>>>(qkv_w, wh, QKVC * DIM);
    gemm_f16_wmma_kernel<<<gemm_blocks(MROWS, QKVC), 128, 0, stream>>>(
        xh, wh, qkv_b, qkv, MROWS, QKVC, DIM, DIM, DIM, QKVC);

    // 2) clustering prep
    extract_q_kernel<<<(BH * NTOK) / 4, 128, 0, stream>>>(qkv, qpre, qn);
    cluster_init_kernel<<<BH * NW, 32, 0, stream>>>(qkv, qavg);

    // 3) two soft-clustering iterations (per bh, sequential on stream)
    for (int it = 0; it < 2; ++it) {
        for (int bh = 0; bh < BH; ++bh) {
            gemm_f16_wmma_kernel<<<gemm_blocks(NTOK, NW), 128, 0, stream>>>(
                qn + (size_t)bh * NTOK * HD, qavg + (size_t)bh * NW * HD, nullptr,
                scores, NTOK, NW, HD, HD, HD, NW);
            softmax_rows_kernel<<<NTOK / 8, 256, 0, stream>>>(scores, probs_t);
            gemm_f16_wmma_kernel<<<gemm_blocks(HD, NW), 128, 0, stream>>>(
                qpre + (size_t)bh * HD * NTOK, probs_t, nullptr,
                qacc, HD, NW, NTOK, NTOK, NTOK, NW);
            norm_qavg_kernel<<<(NW + 63) / 64, 64, 0, stream>>>(
                qacc, qavg + (size_t)bh * NW * HD);
        }
    }

    // 4) final scores -> ratio
    for (int bh = 0; bh < BH; ++bh) {
        gemm_f16_wmma_kernel<<<gemm_blocks(NTOK, NW), 128, 0, stream>>>(
            qn + (size_t)bh * NTOK * HD, qavg + (size_t)bh * NW * HD, nullptr,
            scores, NTOK, NW, HD, HD, HD, NW);
        ratio_kernel<<<NTOK / 256, 256, 0, stream>>>(scores, ratio + (size_t)bh * NTOK);
    }

    // 5) stable ascending argsort + inverse permutation
    sort_kernel<<<BH, 1024, (size_t)NTOK * 8, stream>>>(ratio, qidx);
    invperm_kernel<<<(BH * NTOK) / 256, 256, 0, stream>>>(qidx, qrev);

    // 6) gather permuted q/k/v windows (f16)
    {
        const long long tot = 3LL * NWIN * WS2 * DIM;
        gather_windows_kernel<<<(int)((tot + 255) / 256), 256, 0, stream>>>(qkv, qidx, win);
    }

    // 7) windowed attention (WMMA)
    attn_win_kernel<<<NWIN, 128, 0, stream>>>(
        win,
        win + (size_t)NWIN * WS2 * DIM,
        win + 2 * (size_t)NWIN * WS2 * DIM,
        attn);

    // 8) fused attention-output gather + depthwise conv + add
    conv_merge_kernel<<<(BB * NTOK * DIM) / 256, 256, 0, stream>>>(
        qkv, attn, qrev, conv_w, conv_b, xpre);

    // 9) output projection: out = xpre @ proj_w^T + proj_b
    //    proj_w is (N=192, K=192) row-major == N-major B operand directly.
    cvt_f32_f16_kernel<<<(MROWS * DIM + 255) / 256, 256, 0, stream>>>(xpre, xh, MROWS * DIM);
    cvt_f32_f16_kernel<<<(DIM * DIM + 255) / 256, 256, 0, stream>>>(proj_w, wh, DIM * DIM);
    gemm_f16_wmma_kernel<<<gemm_blocks(MROWS, DIM), 128, 0, stream>>>(
        xh, wh, proj_b, out, MROWS, DIM, DIM, DIM, DIM, DIM);

    (void)in_sizes; (void)n_in; (void)out_size; (void)ws_size;
}
